// RecurrentRGCN_47777216201146
// MI455X (gfx1250) — matrix operand order repaired and checked by reading, not compiled, not run on hardware
//
#include <hip/hip_runtime.h>
#include <math.h>

#define NENT   50000
#define HD     128
#define HD4    32          // float4 per row
#define NREL2  460
#define TSTEPS 8
#define NEDGE  600000
#define HH     (HD * HD)
#define R2H    (NREL2 * HD)
#define SLOPE  0.22916666666666666f

typedef __attribute__((ext_vector_type(16))) __bf16 v16bf;
typedef __attribute__((ext_vector_type(8)))  float  v8f;

// ---------- WMMA operand loaders (contiguous k order, same for A and B) ----------
// A (16x32, 16-bit): lane<16 -> M=lane, K = kb..kb+15;
//                    lane>=16 -> M=lane-16, K = kb+16..kb+31
// Loads are clamped (always-valid address, branchless) and zeroed by select.
__device__ __forceinline__ void loadA(const float* __restrict__ A, int M, int r0, int kb,
                                      int half, int lr, v16bf& hi, v16bf& lo) {
  int row = r0 + lr;
  bool valid = row < M;
  int rowc = valid ? row : (M - 1);
  const float* Ar = A + (size_t)rowc * HD + kb + half * 16;
  float4 a0 = ((const float4*)Ar)[0];
  float4 a1 = ((const float4*)Ar)[1];
  float4 a2 = ((const float4*)Ar)[2];
  float4 a3 = ((const float4*)Ar)[3];
  float av[16] = {a0.x, a0.y, a0.z, a0.w, a1.x, a1.y, a1.z, a1.w,
                  a2.x, a2.y, a2.z, a2.w, a3.x, a3.y, a3.z, a3.w};
#pragma unroll
  for (int i = 0; i < 16; ++i) {
    float a = valid ? av[i] : 0.0f;
    __bf16 ah = (__bf16)a;
    hi[i] = ah;
    lo[i] = (__bf16)(a - (float)ah);
  }
}
// B operand from TRANSPOSED weights Bt[col][k]:
// lane<16 -> N=lane, K = kb+0..15 ; lane>=16 -> N=lane-16, K = kb+16..31
// Contiguous 16-float run per lane -> 4x global_load_b128.
__device__ __forceinline__ void loadB(const float* __restrict__ Bt, int c0, int kb,
                                      int half, int lr, v16bf& hi, v16bf& lo) {
  int col = c0 + lr;
  const float* Bc = Bt + (size_t)col * HD + kb + half * 16;
  float4 b0 = ((const float4*)Bc)[0];
  float4 b1 = ((const float4*)Bc)[1];
  float4 b2 = ((const float4*)Bc)[2];
  float4 b3 = ((const float4*)Bc)[3];
  float bv[16] = {b0.x, b0.y, b0.z, b0.w, b1.x, b1.y, b1.z, b1.w,
                  b2.x, b2.y, b2.z, b2.w, b3.x, b3.y, b3.z, b3.w};
#pragma unroll
  for (int i = 0; i < 16; ++i) {
    __bf16 bh = (__bf16)bv[i];
    hi[i] = bh;
    lo[i] = (__bf16)(bv[i] - (float)bh);
  }
}

// error-compensated bf16 product: C += Ahi*Bhi + Ahi*Blo + Alo*Bhi
__device__ __forceinline__ v8f wmma_split(v16bf ah, v16bf al, v16bf bh, v16bf bl, v8f c) {
  c = __builtin_amdgcn_wmma_f32_16x16x32_bf16(false, ah, false, bh, (short)0, c, false, false);
  c = __builtin_amdgcn_wmma_f32_16x16x32_bf16(false, ah, false, bl, (short)0, c, false, false);
  c = __builtin_amdgcn_wmma_f32_16x16x32_bf16(false, al, false, bh, (short)0, c, false, false);
  return c;
}

// ---------- weight transpose (once per call; weights are loop-invariant) ----------
__global__ void k_transpose(const float* __restrict__ B, float* __restrict__ Bt) {
  int idx = blockIdx.x * blockDim.x + threadIdx.x;
  if (idx >= HH) return;
  int k = idx >> 7, c = idx & (HD - 1);
  Bt[(size_t)c * HD + k] = B[idx];
}

// ---------- C = A[MxHD] @ B[HDxHD] (B given transposed), one wave per 16x16 tile ----------
__global__ void k_gemm(const float* __restrict__ A, const float* __restrict__ Bt,
                       float* __restrict__ C, int M) {
  int wid = blockIdx.x * (blockDim.x >> 5) + (threadIdx.x >> 5);
  int tilesM = (M + 15) >> 4;
  if (wid >= tilesM * 8) return;                 // wave-uniform exit (EXEC stays full)
  int tm = wid >> 3, tn = wid & 7;
  int lane = threadIdx.x & 31, half = lane >> 4, lr = lane & 15;
  int r0 = tm << 4, c0 = tn << 4;
  v8f acc = {};
#pragma unroll
  for (int kb = 0; kb < HD; kb += 32) {
    v16bf ah, al, bh, bl;
    loadA(A, M, r0, kb, half, lr, ah, al);
    loadB(Bt, c0, kb, half, lr, bh, bl);
    acc = wmma_split(ah, al, bh, bl, acc);
  }
#pragma unroll
  for (int j = 0; j < 8; ++j) {                  // C layout: elem j -> row r0+8*half+j
    int row = r0 + (half << 3) + j;
    if (row < M) C[(size_t)row * HD + c0 + lr] = acc[j];
  }
}

// ---------- loop = (indeg>0) ? A@Bl : A@Be  (A registers reused for both B's) ----------
__global__ void k_gemm_sel(const float* __restrict__ A, const float* __restrict__ Blt,
                           const float* __restrict__ Bet, const float* __restrict__ indeg,
                           float* __restrict__ C, int M) {
  int wid = blockIdx.x * (blockDim.x >> 5) + (threadIdx.x >> 5);
  int tilesM = (M + 15) >> 4;
  if (wid >= tilesM * 8) return;
  int tm = wid >> 3, tn = wid & 7;
  int lane = threadIdx.x & 31, half = lane >> 4, lr = lane & 15;
  int r0 = tm << 4, c0 = tn << 4;
  v8f accL = {}, accE = {};
#pragma unroll
  for (int kb = 0; kb < HD; kb += 32) {
    v16bf ah, al, bh, bl;
    loadA(A, M, r0, kb, half, lr, ah, al);
    loadB(Blt, c0, kb, half, lr, bh, bl);
    accL = wmma_split(ah, al, bh, bl, accL);
    loadB(Bet, c0, kb, half, lr, bh, bl);
    accE = wmma_split(ah, al, bh, bl, accE);
  }
#pragma unroll
  for (int j = 0; j < 8; ++j) {
    int row = r0 + (half << 3) + j;
    if (row < M) {
      float sel = (indeg[row] > 0.0f) ? accL[j] : accE[j];
      C[(size_t)row * HD + c0 + lr] = sel;
    }
  }
}

// ---------- in-degree count ----------
__global__ void k_indeg(const int* __restrict__ dst, float* __restrict__ indeg, int ne) {
  int e = blockIdx.x * blockDim.x + threadIdx.x;
  if (e < ne) atomicAdd(&indeg[dst[e]], 1.0f);
}

// ---------- edge scatter: agg[d] += hW[s] + relW[t], one wave per edge ----------
__global__ void k_scatter(const int* __restrict__ src, const int* __restrict__ dst,
                          const int* __restrict__ ety, const float* __restrict__ hW,
                          const float* __restrict__ relW, float* __restrict__ agg, int ne) {
  int e = blockIdx.x * (blockDim.x >> 5) + (threadIdx.x >> 5);
  if (e >= ne) return;
  int lane = threadIdx.x & 31;
  int s = src[e], d = dst[e], t = ety[e];
  float4 hv = ((const float4*)(hW + (size_t)s * HD))[lane];
  float4 rv = ((const float4*)(relW + (size_t)t * HD))[lane];
  float* ad = agg + (size_t)d * HD + lane * 4;
  atomicAdd(ad + 0, hv.x + rv.x);
  atomicAdd(ad + 1, hv.y + rv.y);
  atomicAdd(ad + 2, hv.z + rv.z);
  atomicAdd(ad + 3, hv.w + rv.w);
}

// ---------- layer finish: out = rrelu(agg*norm + loop) ----------
__global__ void k_finish(const float* __restrict__ agg, const float* __restrict__ loopb,
                         const float* __restrict__ indeg, float* __restrict__ out) {
  int idx = blockIdx.x * blockDim.x + threadIdx.x;   // one float4 each
  if (idx >= NENT * HD4) return;
  int n = idx >> 5;
  float ind = indeg[n];
  float nrm = 1.0f / fmaxf(ind, 1.0f);
  float4 a = ((const float4*)agg)[idx];
  float4 l = ((const float4*)loopb)[idx];
  float4 z = {a.x * nrm + l.x, a.y * nrm + l.y, a.z * nrm + l.z, a.w * nrm + l.w};
  float4 r = {z.x >= 0.f ? z.x : SLOPE * z.x, z.y >= 0.f ? z.y : SLOPE * z.y,
              z.z >= 0.f ? z.z : SLOPE * z.z, z.w >= 0.f ? z.w : SLOPE * z.w};
  ((float4*)out)[idx] = r;
}

// ---------- h0 = l2norm(emb_ent), one wave per row ----------
__global__ void k_init_h0(const float* __restrict__ emb, float* __restrict__ h,
                          float* __restrict__ out0) {
  int w = blockIdx.x * (blockDim.x >> 5) + (threadIdx.x >> 5);
  if (w >= NENT) return;
  int lane = threadIdx.x & 31;
  float4 v = ((const float4*)(emb + (size_t)w * HD))[lane];
  float ss = v.x * v.x + v.y * v.y + v.z * v.z + v.w * v.w;
#pragma unroll
  for (int m = 16; m; m >>= 1) ss += __shfl_xor(ss, m, 32);
  float inv = 1.0f / fmaxf(sqrtf(ss), 1e-12f);
  float4 r = {v.x * inv, v.y * inv, v.z * inv, v.w * inv};
  ((float4*)(h + (size_t)w * HD))[lane] = r;
  ((float4*)(out0 + (size_t)w * HD))[lane] = r;
}

// ---------- step combine: h_new = sig(twlin+b)*l2norm(c2) + (1-sig)*h ----------
__global__ void k_combine(const float* __restrict__ c2, const float* __restrict__ twlin,
                          const float* __restrict__ tgb, float* __restrict__ h,
                          float* __restrict__ outT) {
  int w = blockIdx.x * (blockDim.x >> 5) + (threadIdx.x >> 5);
  if (w >= NENT) return;
  int lane = threadIdx.x & 31;
  size_t base = (size_t)w * HD;
  float4 v = ((const float4*)(c2 + base))[lane];
  float ss = v.x * v.x + v.y * v.y + v.z * v.z + v.w * v.w;
#pragma unroll
  for (int m = 16; m; m >>= 1) ss += __shfl_xor(ss, m, 32);
  float inv = 1.0f / fmaxf(sqrtf(ss), 1e-12f);
  float4 cur = {v.x * inv, v.y * inv, v.z * inv, v.w * inv};
  float4 tl = ((const float4*)(twlin + base))[lane];
  float4 bb = ((const float4*)tgb)[lane];
  float4 hv = ((const float4*)(h + base))[lane];
  float tx = 1.0f / (1.0f + expf(-(tl.x + bb.x)));
  float ty = 1.0f / (1.0f + expf(-(tl.y + bb.y)));
  float tz = 1.0f / (1.0f + expf(-(tl.z + bb.z)));
  float tw = 1.0f / (1.0f + expf(-(tl.w + bb.w)));
  float4 hn = {tx * cur.x + (1.0f - tx) * hv.x, ty * cur.y + (1.0f - ty) * hv.y,
               tz * cur.z + (1.0f - tz) * hv.z, tw * cur.w + (1.0f - tw) * hv.w};
  ((float4*)(h + base))[lane] = hn;
  ((float4*)(outT + base))[lane] = hn;
}

static inline int cdiv(int a, int b) { return (a + b - 1) / b; }

extern "C" void kernel_launch(void* const* d_in, const int* in_sizes, int n_in,
                              void* d_out, int out_size, void* d_ws, size_t ws_size,
                              hipStream_t stream) {
  (void)in_sizes; (void)n_in; (void)out_size; (void)ws_size;
  const float* emb_ent  = (const float*)d_in[0];
  const float* emb_rel  = (const float*)d_in[1];
  const float* w_neigh  = (const float*)d_in[2];   // [2,HD,HD]
  const float* w_loop   = (const float*)d_in[3];
  const float* w_evolve = (const float*)d_in[4];
  const float* tg_w     = (const float*)d_in[5];
  const float* tg_b     = (const float*)d_in[6];
  const int*   src      = (const int*)d_in[7];     // [T,E]
  const int*   dst      = (const int*)d_in[8];
  const int*   ety      = (const int*)d_in[9];
  float* out = (float*)d_out;

  const size_t NH = (size_t)NENT * HD;
  float* ws     = (float*)d_ws;
  float* buf_h  = ws;                 // persistent hidden state
  float* buf_a  = buf_h + NH;
  float* buf_b  = buf_a + NH;
  float* buf_c  = buf_b + NH;
  float* relW   = buf_c + NH;         // [2, NREL2, HD]
  float* indeg  = relW + 2 * (size_t)R2H;
  float* wT     = indeg + NENT;       // 7 transposed HDxHD weights
  float* wnT0 = wT;            float* wnT1 = wT + 1 * (size_t)HH;
  float* wlT0 = wT + 2 * (size_t)HH; float* wlT1 = wT + 3 * (size_t)HH;
  float* weT0 = wT + 4 * (size_t)HH; float* weT1 = wT + 5 * (size_t)HH;
  float* tgT  = wT + 6 * (size_t)HH;

  // GEMM grids: 256 threads = 8 waves; one wave per 16x16 tile, 8 col tiles per row strip
  const int gemmBlkN = cdiv(cdiv(NENT, 16) * 8, 8);     // M = NENT
  const int gemmBlkR = cdiv(cdiv(NREL2, 16) * 8, 8);    // M = NREL2
  const int trBlk = cdiv(HH, 256);

  // transpose all weights once (loop-invariant)
  k_transpose<<<trBlk, 256, 0, stream>>>(w_neigh, wnT0);
  k_transpose<<<trBlk, 256, 0, stream>>>(w_neigh + HH, wnT1);
  k_transpose<<<trBlk, 256, 0, stream>>>(w_loop, wlT0);
  k_transpose<<<trBlk, 256, 0, stream>>>(w_loop + HH, wlT1);
  k_transpose<<<trBlk, 256, 0, stream>>>(w_evolve, weT0);
  k_transpose<<<trBlk, 256, 0, stream>>>(w_evolve + HH, weT1);
  k_transpose<<<trBlk, 256, 0, stream>>>(tg_w, tgT);

  // h0 = l2norm(emb_ent) -> state + output slice 0
  k_init_h0<<<cdiv(NENT, 8), 256, 0, stream>>>(emb_ent, buf_h, out);

  // relW[l] = emb_rel @ w_neigh[l]  (time-invariant, hoisted out of the loop)
  k_gemm<<<gemmBlkR, 256, 0, stream>>>(emb_rel, wnT0, relW, NREL2);
  k_gemm<<<gemmBlkR, 256, 0, stream>>>(emb_rel, wnT1, relW + R2H, NREL2);

  // second output: emb_rel passthrough
  hipMemcpyAsync(out + 9 * NH, emb_rel, (size_t)R2H * sizeof(float),
                 hipMemcpyDeviceToDevice, stream);

  for (int t = 0; t < TSTEPS; ++t) {
    const int* s = src + (size_t)t * NEDGE;
    const int* d = dst + (size_t)t * NEDGE;
    const int* e = ety + (size_t)t * NEDGE;

    hipMemsetAsync(indeg, 0, NENT * sizeof(float), stream);
    k_indeg<<<cdiv(NEDGE, 256), 256, 0, stream>>>(d, indeg, NEDGE);

    // ---- layer 0: x = buf_h ----
    k_gemm<<<gemmBlkN, 256, 0, stream>>>(buf_h, wnT0, buf_a, NENT);             // hW
    hipMemsetAsync(buf_b, 0, NH * sizeof(float), stream);                       // agg = 0
    k_scatter<<<cdiv(NEDGE, 8), 256, 0, stream>>>(s, d, e, buf_a, relW, buf_b, NEDGE);
    k_gemm_sel<<<gemmBlkN, 256, 0, stream>>>(buf_h, wlT0, weT0, indeg, buf_c, NENT);
    k_finish<<<cdiv(NENT * HD4, 256), 256, 0, stream>>>(buf_b, buf_c, indeg, buf_a);

    // ---- layer 1: x = buf_a ----
    k_gemm<<<gemmBlkN, 256, 0, stream>>>(buf_a, wnT1, buf_b, NENT);             // hW
    hipMemsetAsync(buf_c, 0, NH * sizeof(float), stream);                       // agg = 0
    k_scatter<<<cdiv(NEDGE, 8), 256, 0, stream>>>(s, d, e, buf_b, relW + R2H, buf_c, NEDGE);
    k_gemm_sel<<<gemmBlkN, 256, 0, stream>>>(buf_a, wlT1, weT1, indeg, buf_b, NENT);
    k_finish<<<cdiv(NENT * HD4, 256), 256, 0, stream>>>(buf_c, buf_b, indeg, buf_b); // in-place

    // ---- time gate + combine ----
    k_gemm<<<gemmBlkN, 256, 0, stream>>>(buf_h, tgT, buf_c, NENT);              // twlin
    k_combine<<<cdiv(NENT, 8), 256, 0, stream>>>(buf_b, buf_c, tg_b, buf_h,
                                                 out + (size_t)(t + 1) * NH);
  }
}
